// LSTMDecoder_65266323030442
// MI455X (gfx1250) — matrix operand order repaired and checked by reading, not compiled, not run on hardware
//
#include <hip/hip_runtime.h>
#include <hip/hip_bf16.h>
#include <stdint.h>

// ---------------------------------------------------------------------------
// Types for CDNA5 WMMA (wave32): 16x16x32 bf16 -> f32
// ---------------------------------------------------------------------------
typedef __bf16 bf16_t;
typedef __attribute__((ext_vector_type(16))) __bf16 v16bf;
typedef __attribute__((ext_vector_type(8)))  __bf16 v8bf;
typedef __attribute__((ext_vector_type(8)))  float  v8f;

#define DEV __device__ __forceinline__

static constexpr int Bb = 64, Nn = 196, Aa = 512, Hh = 512, Ee = 512, Vv = 10000;
static constexpr int Tt = 32, STEPS = 31;

DEV bf16_t f2bf(float f) { return (bf16_t)f; }

// A fragment: 16x32 bf16 tile, row-major source, contiguous K per lane half (same order as load_bt_frag):
// lanes 0-15: row m0+lane, K = k0..k0+15
// lanes 16-31: row m0+lane-16, K = k0+16..k0+31
DEV v16bf load_a_frag(const bf16_t* __restrict__ A, int lda, int m0, int k0, int lane) {
    const bf16_t* p = A + (size_t)(m0 + (lane & 15)) * lda + k0 + ((lane >> 4) << 4);
    v8bf lo = *(const v8bf*)p;          // 16B vector load
    v8bf hi = *(const v8bf*)(p + 8);    // 16B vector load
    v16bf r;
#pragma unroll
    for (int i = 0; i < 8; ++i) { r[i] = lo[i]; r[i + 8] = hi[i]; }
    return r;
}

// B fragment for C = A * W^T with W row-major (N x K):
// lanes 0-15: col n0+lane, K = k0..k0+15 ; lanes 16-31: col n0+lane-16, K = k0+16..k0+31
// -> 16 contiguous bf16 per lane, one 32B vector load.
DEV v16bf load_bt_frag(const bf16_t* __restrict__ W, int ldw, int n0, int k0, int lane) {
    const bf16_t* p = W + (size_t)(n0 + (lane & 15)) * ldw + k0 + ((lane >> 4) << 4);
    return *(const v16bf*)p;
}

DEV v8f wmma_bf16(v16bf a, v16bf b, v8f c) {
    return __builtin_amdgcn_wmma_f32_16x16x32_bf16(false, a, false, b, (short)0, c, false, false);
}

DEV float sigmoidf_(float x) { return 1.0f / (1.0f + __expf(-x)); }

// ---------------------------------------------------------------------------
// Setup kernels
// ---------------------------------------------------------------------------
__global__ void convert_f32_bf16_kernel(const float* __restrict__ src,
                                        bf16_t* __restrict__ dst, int n) {
    int i = blockIdx.x * blockDim.x + threadIdx.x;
    if (i < n) dst[i] = f2bf(src[i]);
}

__global__ void init_state_kernel(const float* __restrict__ h0, const float* __restrict__ c0,
                                  float* __restrict__ h, float* __restrict__ c,
                                  bf16_t* __restrict__ hbf) {
    int i = blockIdx.x * blockDim.x + threadIdx.x;   // 2*64*512
    float hv = h0[i];
    h[i] = hv; c[i] = c0[i]; hbf[i] = f2bf(hv);
}

// fm_proj[b,n] = dot(feature_map[b,n,:], W_attn[0,:A])
__global__ void fmproj_kernel(const float* __restrict__ fm, const float* __restrict__ Wattn,
                              float* __restrict__ fm_proj) {
    __shared__ float red[256];
    int o = blockIdx.x;                 // b*196+n
    int tid = threadIdx.x;
    const float* row = fm + (size_t)o * Aa;
    float p = row[tid] * Wattn[tid] + row[tid + 256] * Wattn[tid + 256];
    red[tid] = p; __syncthreads();
    for (int s = 128; s > 0; s >>= 1) { if (tid < s) red[tid] += red[tid + s]; __syncthreads(); }
    if (tid == 0) fm_proj[o] = red[0];
}

// ---------------------------------------------------------------------------
// Per-step kernels
// ---------------------------------------------------------------------------
__global__ void embed_kernel(const float* __restrict__ emb, const int* __restrict__ caps,
                             int t, bf16_t* __restrict__ xbf) {
    int idx = blockIdx.x * blockDim.x + threadIdx.x;   // 64*512
    int b = idx >> 9, e = idx & 511;
    int tok = caps[b * Tt + t];
    xbf[idx] = f2bf(emb[(size_t)tok * Ee + e]);
}

// gates = act( X @ Wih^T + Hprev @ Whh^T + bih + bhh ), X:64x512, Hprev:64x512
// 128 waves; each wave: 1 M-tile x 4 N-tiles (A-fragment reused 4x per K step).
__global__ void gates_wmma_kernel(const bf16_t* __restrict__ X, const bf16_t* __restrict__ Hp,
                                  const bf16_t* __restrict__ Wih, const bf16_t* __restrict__ Whh,
                                  const float* __restrict__ bih, const float* __restrict__ bhh,
                                  float* __restrict__ gates) {
    int lane = threadIdx.x & 31;
    int wid  = (blockIdx.x * blockDim.x + threadIdx.x) >> 5;   // 0..127
    int n0 = (wid & 31) << 6;          // 4 tiles of 16 -> 64 cols
    int m0 = (wid >> 5) << 4;
    v8f acc[4] = {};
#pragma unroll 2
    for (int k0 = 0; k0 < 512; k0 += 32) {
        v16bf a = load_a_frag(X, 512, m0, k0, lane);
#pragma unroll
        for (int j = 0; j < 4; ++j)
            acc[j] = wmma_bf16(a, load_bt_frag(Wih, 512, n0 + 16 * j, k0, lane), acc[j]);
    }
#pragma unroll 2
    for (int k0 = 0; k0 < 512; k0 += 32) {
        v16bf a = load_a_frag(Hp, 512, m0, k0, lane);
#pragma unroll
        for (int j = 0; j < 4; ++j)
            acc[j] = wmma_bf16(a, load_bt_frag(Whh, 512, n0 + 16 * j, k0, lane), acc[j]);
    }
    int mbase = m0 + ((lane >> 4) << 3);
#pragma unroll
    for (int j = 0; j < 4; ++j) {
        int n = n0 + 16 * j + (lane & 15);
        float bsum = bih[n] + bhh[n];
        int gate = n >> 9;             // 0=i 1=f 2=g 3=o
#pragma unroll
        for (int r = 0; r < 8; ++r) {
            float g = acc[j][r] + bsum;
            float v = (gate == 2) ? tanhf(g) : sigmoidf_(g);
            gates[(size_t)(mbase + r) * 2048 + n] = v;
        }
    }
}

__global__ void lstm_cell_kernel(const float* __restrict__ gates, float* __restrict__ c,
                                 float* __restrict__ h, bf16_t* __restrict__ hbf) {
    int idx = blockIdx.x * blockDim.x + threadIdx.x;   // 64*512
    int m = idx >> 9, j = idx & 511;
    const float* gr = gates + (size_t)m * 2048;
    float gi = gr[j], gf = gr[j + 512], gg = gr[j + 1024], go = gr[j + 1536];
    float c2 = gf * c[idx] + gi * gg;
    float h2 = go * tanhf(c2);
    c[idx] = c2; h[idx] = h2; hbf[idx] = f2bf(h2);
}

// attention: scores = fm_proj + hid.Wa_h + b ; softmax over N ; ctx = attn @ fm
__global__ void attn_kernel(const float* __restrict__ fm, const float* __restrict__ fm_proj,
                            const float* __restrict__ hid, const float* __restrict__ Wattn,
                            const float* __restrict__ battn, bf16_t* __restrict__ ctxbf) {
    __shared__ float red[256];
    __shared__ float attn[Nn];
    int b = blockIdx.x, tid = threadIdx.x;
    // hdot = dot(hid[b], Wa_h)
    float p = hid[b * Hh + tid] * Wattn[Aa + tid] + hid[b * Hh + 256 + tid] * Wattn[Aa + 256 + tid];
    red[tid] = p; __syncthreads();
    for (int s = 128; s > 0; s >>= 1) { if (tid < s) red[tid] += red[tid + s]; __syncthreads(); }
    float hdot = red[0] + battn[0];
    __syncthreads();
    float sc = -3.4e38f;
    if (tid < Nn) sc = fm_proj[b * Nn + tid] + hdot;
    red[tid] = sc; __syncthreads();
    for (int s = 128; s > 0; s >>= 1) { if (tid < s) red[tid] = fmaxf(red[tid], red[tid + s]); __syncthreads(); }
    float mx = red[0]; __syncthreads();
    float e = 0.f;
    if (tid < Nn) e = __expf(sc - mx);
    red[tid] = e; __syncthreads();
    for (int s = 128; s > 0; s >>= 1) { if (tid < s) red[tid] += red[tid + s]; __syncthreads(); }
    float inv = 1.0f / red[0]; __syncthreads();
    if (tid < Nn) attn[tid] = e * inv;
    __syncthreads();
    for (int a = tid; a < Aa; a += 256) {
        const float* fmb = fm + ((size_t)b * Nn) * Aa + a;
        float s = 0.f;
#pragma unroll 4
        for (int n = 0; n < Nn; ++n) s += attn[n] * fmb[(size_t)n * Aa];
        ctxbf[b * Aa + a] = f2bf(s);
    }
}

// pre = x + hid @ Wfch^T + ctx @ Wfcctx^T + biases -> staged bf16 at row t*64+b
// 32 waves; each wave: 1 M-tile x 4 N-tiles.
__global__ void pre_wmma_kernel(const bf16_t* __restrict__ hidbf, const bf16_t* __restrict__ ctxbf,
                                const bf16_t* __restrict__ Wfch, const bf16_t* __restrict__ Wfcctx,
                                const float* __restrict__ bfch, const float* __restrict__ bfcctx,
                                const float* __restrict__ emb, const int* __restrict__ caps,
                                int t, bf16_t* __restrict__ pre_all) {
    int lane = threadIdx.x & 31;
    int wid  = (blockIdx.x * blockDim.x + threadIdx.x) >> 5;   // 0..31
    int n0 = (wid & 7) << 6;
    int m0 = (wid >> 3) << 4;
    v8f acc[4] = {};
#pragma unroll 2
    for (int k0 = 0; k0 < 512; k0 += 32) {
        v16bf a = load_a_frag(hidbf, 512, m0, k0, lane);
#pragma unroll
        for (int j = 0; j < 4; ++j)
            acc[j] = wmma_bf16(a, load_bt_frag(Wfch, 512, n0 + 16 * j, k0, lane), acc[j]);
    }
#pragma unroll 2
    for (int k0 = 0; k0 < 512; k0 += 32) {
        v16bf a = load_a_frag(ctxbf, 512, m0, k0, lane);
#pragma unroll
        for (int j = 0; j < 4; ++j)
            acc[j] = wmma_bf16(a, load_bt_frag(Wfcctx, 512, n0 + 16 * j, k0, lane), acc[j]);
    }
    int mbase = m0 + ((lane >> 4) << 3);
#pragma unroll
    for (int j = 0; j < 4; ++j) {
        int n = n0 + 16 * j + (lane & 15);
        float bias = bfch[n] + bfcctx[n];
#pragma unroll
        for (int r = 0; r < 8; ++r) {
            int m = mbase + r;
            int tok = caps[m * Tt + t];
            float v = acc[j][r] + bias + emb[(size_t)tok * Ee + n];
            pre_all[((size_t)t * Bb + m) * Ee + n] = f2bf(v);
        }
    }
}

// ---------------------------------------------------------------------------
// Batched classifier: logits[1984, 10000] = Pre @ Wclsf^T + b
// Each wave computes a 2(M) x 5(N) block of 16x16 tiles: per K step,
// 2 A-fragments + 5 B-fragments feed 10 WMMAs (1.4 b128 loads / WMMA).
// Tile grid: 62 M-blocks x 125 N-blocks = 7750 waves (exact).
// ---------------------------------------------------------------------------
__global__ void logits_wmma_kernel(const bf16_t* __restrict__ pre_all,
                                   const bf16_t* __restrict__ Wclsf,
                                   const float* __restrict__ bclsf,
                                   float* __restrict__ out) {
    int lane = threadIdx.x & 31;
    int wid  = (blockIdx.x * blockDim.x + threadIdx.x) >> 5;   // 0..7749
    int n_blk = wid % 125, m_blk = wid / 125;
    int n0 = n_blk * 80;               // 5 tiles of 16
    int m0 = m_blk * 32;               // 2 tiles of 16
    v8f acc[2][5] = {};
    for (int k0 = 0; k0 < 512; k0 += 32) {
        if (k0 + 64 < 512)             // warm L2/L0 two K-steps ahead
            __builtin_prefetch(Wclsf + (size_t)(n0 + (lane & 15)) * 512 + k0 + 64, 0, 3);
        v16bf a0 = load_a_frag(pre_all, 512, m0, k0, lane);
        v16bf a1 = load_a_frag(pre_all, 512, m0 + 16, k0, lane);
#pragma unroll
        for (int j = 0; j < 5; ++j) {
            v16bf bfrag = load_bt_frag(Wclsf, 512, n0 + 16 * j, k0, lane);
            acc[0][j] = wmma_bf16(a0, bfrag, acc[0][j]);
            acc[1][j] = wmma_bf16(a1, bfrag, acc[1][j]);
        }
    }
#pragma unroll
    for (int i = 0; i < 2; ++i) {
        int mbase = m0 + 16 * i + ((lane >> 4) << 3);
#pragma unroll
        for (int j = 0; j < 5; ++j) {
            int n = n0 + 16 * j + (lane & 15);
            float bias = bclsf[n];
#pragma unroll
            for (int r = 0; r < 8; ++r) {
                int row = mbase + r;    // row = t*64 + b
                int t = row >> 6, b = row & 63;
                out[((size_t)(b * STEPS + t)) * Vv + n] = acc[i][j][r] + bias;
            }
        }
    }
}

__global__ void argmax_kernel(const float* __restrict__ logits, float* __restrict__ sampled) {
    __shared__ float vmax[256];
    __shared__ int   vidx[256];
    int row = blockIdx.x;               // b*31 + t
    int tid = threadIdx.x;
    const float* lr = logits + (size_t)row * Vv;
    float best = -3.4e38f; int bi = 0;
    for (int n = tid; n < Vv; n += 256) {
        float v = lr[n];
        if (v > best) { best = v; bi = n; }
    }
    vmax[tid] = best; vidx[tid] = bi; __syncthreads();
    for (int s = 128; s > 0; s >>= 1) {
        if (tid < s) {
            if (vmax[tid + s] > vmax[tid] ||
                (vmax[tid + s] == vmax[tid] && vidx[tid + s] < vidx[tid])) {
                vmax[tid] = vmax[tid + s]; vidx[tid] = vidx[tid + s];
            }
        }
        __syncthreads();
    }
    if (tid == 0) sampled[row] = (float)vidx[0];
}

// ---------------------------------------------------------------------------
// Host orchestration
// ---------------------------------------------------------------------------
extern "C" void kernel_launch(void* const* d_in, const int* in_sizes, int n_in,
                              void* d_out, int out_size, void* d_ws, size_t ws_size,
                              hipStream_t stream) {
    (void)in_sizes; (void)n_in; (void)out_size; (void)ws_size;
    const float* fm     = (const float*)d_in[0];
    const float* h0_in  = (const float*)d_in[1];
    const float* c0_in  = (const float*)d_in[2];
    const int*   caps   = (const int*)d_in[3];
    const float* emb    = (const float*)d_in[5];
    const float* Wih0   = (const float*)d_in[6];
    const float* Whh0   = (const float*)d_in[7];
    const float* bih0   = (const float*)d_in[8];
    const float* bhh0   = (const float*)d_in[9];
    const float* Wih1   = (const float*)d_in[10];
    const float* Whh1   = (const float*)d_in[11];
    const float* bih1   = (const float*)d_in[12];
    const float* bhh1   = (const float*)d_in[13];
    const float* Wattn  = (const float*)d_in[14];
    const float* battn  = (const float*)d_in[15];
    const float* Wfch   = (const float*)d_in[16];
    const float* bfch   = (const float*)d_in[17];
    const float* Wfcctx = (const float*)d_in[18];
    const float* bfcctx = (const float*)d_in[19];
    const float* Wclsf  = (const float*)d_in[20];
    const float* bclsf  = (const float*)d_in[21];

    char* ws = (char*)d_ws;
    size_t off = 0;
    auto alloc = [&](size_t bytes) -> void* {
        void* p = ws + off;
        off = (off + bytes + 255) & ~(size_t)255;
        return p;
    };

    bf16_t* wih0b = (bf16_t*)alloc((size_t)4 * Hh * Ee * 2);
    bf16_t* whh0b = (bf16_t*)alloc((size_t)4 * Hh * Hh * 2);
    bf16_t* wih1b = (bf16_t*)alloc((size_t)4 * Hh * Hh * 2);
    bf16_t* whh1b = (bf16_t*)alloc((size_t)4 * Hh * Hh * 2);
    bf16_t* wfchb = (bf16_t*)alloc((size_t)Ee * Hh * 2);
    bf16_t* wfcxb = (bf16_t*)alloc((size_t)Ee * Aa * 2);
    bf16_t* wclsb = (bf16_t*)alloc((size_t)Vv * Ee * 2);
    float*  fmproj = (float*)alloc((size_t)Bb * Nn * 4);
    float*  hstate = (float*)alloc((size_t)2 * Bb * Hh * 4);
    float*  cstate = (float*)alloc((size_t)2 * Bb * Hh * 4);
    bf16_t* hbf    = (bf16_t*)alloc((size_t)2 * Bb * Hh * 2);
    bf16_t* xbf    = (bf16_t*)alloc((size_t)Bb * Ee * 2);
    float*  gates  = (float*)alloc((size_t)Bb * 4 * Hh * 4);
    bf16_t* ctxbf  = (bf16_t*)alloc((size_t)Bb * Aa * 2);
    bf16_t* preall = (bf16_t*)alloc((size_t)STEPS * Bb * Ee * 2);

    float* out_logits  = (float*)d_out;
    float* out_sampled = out_logits + (size_t)Bb * STEPS * Vv;

    auto cvt = [&](const float* s, bf16_t* d, int n) {
        convert_f32_bf16_kernel<<<(n + 255) / 256, 256, 0, stream>>>(s, d, n);
    };
    cvt(Wih0,  wih0b, 4 * Hh * Ee);
    cvt(Whh0,  whh0b, 4 * Hh * Hh);
    cvt(Wih1,  wih1b, 4 * Hh * Hh);
    cvt(Whh1,  whh1b, 4 * Hh * Hh);
    cvt(Wfch,  wfchb, Ee * Hh);
    cvt(Wfcctx, wfcxb, Ee * Aa);
    cvt(Wclsf, wclsb, Vv * Ee);

    init_state_kernel<<<(2 * Bb * Hh) / 256, 256, 0, stream>>>(h0_in, c0_in, hstate, cstate, hbf);
    fmproj_kernel<<<Bb * Nn, 256, 0, stream>>>(fm, Wattn, fmproj);

    bf16_t* h0bf = hbf;                 // layer-0 h (bf16)
    bf16_t* h1bf = hbf + Bb * Hh;       // layer-1 h (bf16) = hid
    float*  h0f  = hstate;
    float*  h1f  = hstate + Bb * Hh;
    float*  c0f  = cstate;
    float*  c1f  = cstate + Bb * Hh;

    for (int t = 0; t < STEPS; ++t) {
        embed_kernel<<<(Bb * Ee) / 256, 256, 0, stream>>>(emb, caps, t, xbf);
        // layer 0: 128 waves / 8 per block
        gates_wmma_kernel<<<16, 256, 0, stream>>>(xbf, h0bf, wih0b, whh0b, bih0, bhh0, gates);
        lstm_cell_kernel<<<(Bb * Hh) / 256, 256, 0, stream>>>(gates, c0f, h0f, h0bf);
        // layer 1
        gates_wmma_kernel<<<16, 256, 0, stream>>>(h0bf, h1bf, wih1b, whh1b, bih1, bhh1, gates);
        lstm_cell_kernel<<<(Bb * Hh) / 256, 256, 0, stream>>>(gates, c1f, h1f, h1bf);
        // attention + context
        attn_kernel<<<Bb, 256, 0, stream>>>(fm, fmproj, h1f, Wattn, battn, ctxbf);
        // pre projection, staged for the batched classifier (32 waves / 8 per block)
        pre_wmma_kernel<<<4, 256, 0, stream>>>(h1bf, ctxbf, wfchb, wfcxb, bfch, bfcctx,
                                               emb, caps, t, preall);
    }

    // Batched classifier GEMM: 7750 waves, 5 waves per 160-thread block (exact).
    logits_wmma_kernel<<<1550, 160, 0, stream>>>(preall, wclsb, bclsf, out_logits);
    argmax_kernel<<<Bb * STEPS, 256, 0, stream>>>(out_logits, out_sampled);
}
